// ClinicalODERNN_51256139710925
// MI455X (gfx1250) — compile-verified, matrix operand order checked by reading, not run on hardware
//
#include <hip/hip_runtime.h>
#include <hip/hip_bf16.h>
#include <math.h>

// Problem constants (from reference)
#define B_   512
#define S_   128
#define F_   128
#define H_   512
#define H2_  1024
#define H3_  1536

typedef __attribute__((ext_vector_type(16))) __bf16 v16bf;
typedef __attribute__((ext_vector_type(8)))  float  v8f;

// Dormand-Prince tableau (stage-argument coefficients and final weights)
__constant__ float RK_A[5][5] = {
    {0.2f, 0.f, 0.f, 0.f, 0.f},
    {3.f/40.f, 9.f/40.f, 0.f, 0.f, 0.f},
    {44.f/45.f, -56.f/15.f, 32.f/9.f, 0.f, 0.f},
    {19372.f/6561.f, -25360.f/2187.f, 64448.f/6561.f, -212.f/729.f, 0.f},
    {9017.f/3168.f, -355.f/33.f, 46732.f/5247.f, 49.f/176.f, -5103.f/18656.f}};
__constant__ float RK_B[6] = {35.f/384.f, 0.f, 500.f/1113.f, 125.f/192.f,
                              -2187.f/6784.f, 11.f/84.f};

// ---------------------------------------------------------------------------
// Prep kernel 1: per-step effective RK substep size
// dtm[s] = mean_b(t[b,s]-t[b,s-1])/72 ; dt_sub[s] = (dtm>1e-4 ? dtm : 0)/8
// ---------------------------------------------------------------------------
__global__ void compute_dt_kernel(const float* __restrict__ tp,
                                  float* __restrict__ dtsub) {
    int s = blockIdx.x;
    __shared__ float red[256];
    float acc = 0.f;
    if (s > 0)
        for (int b = threadIdx.x; b < B_; b += 256)
            acc += tp[(size_t)b * S_ + s] - tp[(size_t)b * S_ + s - 1];
    red[threadIdx.x] = acc;
    __syncthreads();
    for (int o = 128; o > 0; o >>= 1) {
        if (threadIdx.x < o) red[threadIdx.x] += red[threadIdx.x + o];
        __syncthreads();
    }
    if (threadIdx.x == 0) {
        float dtm = red[0] / (512.f * 72.f);
        dtsub[s] = (dtm > 1e-4f ? dtm : 0.f) / 8.f;
    }
}

// ---------------------------------------------------------------------------
// Prep kernel 2: f32 -> bf16 weight transpose into N-major layout:
//   dst[n*K + k] = bf16(src[k*sK + n*sN])
// so a WMMA B-fragment lane reads 16 contiguous K values for one column n.
// ---------------------------------------------------------------------------
__global__ void conv_w_kernel(const float* __restrict__ src,
                              __bf16* __restrict__ dst,
                              int K, int N, int sK, int sN) {
    int total = K * N;
    for (int i = blockIdx.x * blockDim.x + threadIdx.x; i < total;
         i += gridDim.x * blockDim.x) {
        int n = i / K, k = i % K;
        dst[i] = (__bf16)src[(size_t)k * sK + (size_t)n * sN];
    }
}

// ---------------------------------------------------------------------------
// Main persistent kernel: one workgroup = one 16-row batch tile, full scan.
// 16 waves / 512 threads. Dynamic LDS layout (272 KB):
//   [0      .. 32768 )  h      f32 16x512
//   [32768  .. 65536 )  hacc   f32 16x512      (GRU: r-gate buffer)
//   [65536  .. 229376)  kbuf   f32 5 x 16x512  (GRU: ghb 16x1536 | z | gi_n)
//   [229376 .. 245760)  a_bf   bf16 16x512  (A operand staging)
//   [245760 .. 278528)  act    bf16 16x1024 (layer-1 activations / GRU xb)
// ---------------------------------------------------------------------------
__global__ __launch_bounds__(512)
void ode_rnn_kernel(const float* __restrict__ tp,
                    const float* __restrict__ values,
                    const float* __restrict__ dtsub,
                    const __bf16* __restrict__ W1T,   // [1024][512]
                    const __bf16* __restrict__ W2T,   // [512][1024]
                    const __bf16* __restrict__ WHH,   // [1536][512]
                    const __bf16* __restrict__ WIH,   // [1536][128]
                    const float* __restrict__ b1,
                    const float* __restrict__ b2,
                    const float* __restrict__ b_ih,
                    const float* __restrict__ b_hh,
                    const float* __restrict__ w_ih_full,  // (1536,129) f32
                    float* __restrict__ out) {
    extern __shared__ __align__(256) char smem[];
    float*  h    = (float*)(smem);
    float*  hacc = (float*)(smem + 32768);
    float*  kbuf = (float*)(smem + 65536);
    __bf16* a_bf = (__bf16*)(smem + 229376);
    __bf16* act  = (__bf16*)(smem + 245760);
    // GRU-phase aliases
    float*  rbuf = hacc;                       // 16x512
    float*  ghb  = (float*)(smem + 65536);     // 16x1536
    float*  zbuf = (float*)(smem + 163840);    // 16x512
    float*  ginb = (float*)(smem + 196608);    // 16x512
    __bf16* xb   = (__bf16*)(smem + 245760);   // 16x128 bf16
    __shared__ float s_dti[16];

    const int tid  = threadIdx.x;
    const int wave = tid >> 5;
    const int lane = tid & 31;
    const int bt   = blockIdx.x;        // batch tile (16 rows)
    const int arow = lane & 15;         // fragment row/col within tile
    const int kh   = (lane >> 4) * 16;  // K-half offset per lane
    const int rb   = (lane >> 4) * 8;   // C-fragment row base

    for (int e = tid; e < 16 * H_; e += 512) {
        h[e] = 0.f; hacc[e] = 0.f; a_bf[e] = (__bf16)0.f;
    }
    __syncthreads();

    for (int s = 0; s < S_; ++s) {
        const float dt = dtsub[s];
        if (dt > 0.f) {
            for (int sub = 0; sub < 8; ++sub) {
                for (int stage = 0; stage < 6; ++stage) {
                    __syncthreads();
                    // ---- layer1: act = tanh(arg @ w1 + b1), 16x1024
                    {
                        v8f acc[4] = {};
                        const __bf16* Ap = a_bf + arow * H_ + kh;
                        const __bf16* Bbase = W1T + (size_t)arow * H_ + kh;
                        #pragma unroll 2
                        for (int k0 = 0; k0 < H_; k0 += 32) {
                            v16bf af = *(const v16bf*)(Ap + k0);
                            __builtin_prefetch(Bbase + (size_t)(wave << 4) * H_ + k0 + 64, 0, 3);
                            #pragma unroll
                            for (int i = 0; i < 4; ++i) {
                                int n0 = (wave + 16 * i) << 4;
                                v16bf bfr = *(const v16bf*)(Bbase + (size_t)n0 * H_ + k0);
                                acc[i] = __builtin_amdgcn_wmma_f32_16x16x32_bf16(
                                    false, af, false, bfr, (short)0, acc[i], false, false);
                            }
                        }
                        #pragma unroll
                        for (int i = 0; i < 4; ++i) {
                            int c = ((wave + 16 * i) << 4) + arow;
                            float bias = b1[c];
                            #pragma unroll
                            for (int v = 0; v < 8; ++v) {
                                float val = tanhf(acc[i][v] + bias);
                                act[(size_t)(rb + v) * H2_ + c] = (__bf16)val;
                            }
                        }
                    }
                    __syncthreads();
                    // ---- layer2: k_stage = act @ w2 + b2, 16x512
                    {
                        v8f acc[2] = {};
                        const __bf16* Ap = act + arow * H2_ + kh;
                        const __bf16* Bbase = W2T + (size_t)arow * H2_ + kh;
                        #pragma unroll 2
                        for (int k0 = 0; k0 < H2_; k0 += 32) {
                            v16bf af = *(const v16bf*)(Ap + k0);
                            #pragma unroll
                            for (int i = 0; i < 2; ++i) {
                                int n0 = (wave + 16 * i) << 4;
                                v16bf bfr = *(const v16bf*)(Bbase + (size_t)n0 * H2_ + k0);
                                acc[i] = __builtin_amdgcn_wmma_f32_16x16x32_bf16(
                                    false, af, false, bfr, (short)0, acc[i], false, false);
                            }
                        }
                        #pragma unroll
                        for (int i = 0; i < 2; ++i) {
                            int c = ((wave + 16 * i) << 4) + arow;
                            float bias = b2[c];
                            if (stage < 5) {
                                float* kd = kbuf + stage * 8192;
                                #pragma unroll
                                for (int v = 0; v < 8; ++v)
                                    kd[(rb + v) * H_ + c] = acc[i][v] + bias;
                            } else {  // k6: fold straight into the combination
                                float w = dt * RK_B[5];
                                #pragma unroll
                                for (int v = 0; v < 8; ++v)
                                    hacc[(rb + v) * H_ + c] += w * (acc[i][v] + bias);
                            }
                        }
                    }
                    __syncthreads();
                    if (stage < 5) {
                        // build next stage argument, accumulate final combo
                        for (int e = tid; e < 16 * H_; e += 512) {
                            hacc[e] += dt * RK_B[stage] * kbuf[stage * 8192 + e];
                            float arg = h[e];
                            for (int j = 0; j <= stage; ++j)
                                arg += dt * RK_A[stage][j] * kbuf[j * 8192 + e];
                            a_bf[e] = (__bf16)arg;
                        }
                    } else {
                        for (int e = tid; e < 16 * H_; e += 512) {
                            float hv = h[e] + hacc[e];
                            h[e] = hv; hacc[e] = 0.f; a_bf[e] = (__bf16)hv;
                        }
                    }
                }
            }
        }
        // ---------------- GRU cell ----------------
        __syncthreads();
        for (int e = tid; e < 16 * F_; e += 512) {
            int m = e >> 7, f = e & 127;
            xb[e] = (__bf16)values[((size_t)(bt * 16 + m) * S_ + s) * F_ + f];
        }
        if (tid < 16) {
            int b = bt * 16 + tid;
            s_dti[tid] = (s == 0) ? 0.f
                                  : (tp[(size_t)b * S_ + s] - tp[(size_t)b * S_ + s - 1]);
        }
        __syncthreads();
        // pass A: gh = h @ w_hh^T + b_hh  (16x1536)
        {
            v8f acc[6] = {};
            const __bf16* Ap = a_bf + arow * H_ + kh;
            const __bf16* Bbase = WHH + (size_t)arow * H_ + kh;
            #pragma unroll 1
            for (int k0 = 0; k0 < H_; k0 += 32) {
                v16bf af = *(const v16bf*)(Ap + k0);
                #pragma unroll
                for (int i = 0; i < 6; ++i) {
                    int n0 = (wave + 16 * i) << 4;
                    v16bf bfr = *(const v16bf*)(Bbase + (size_t)n0 * H_ + k0);
                    acc[i] = __builtin_amdgcn_wmma_f32_16x16x32_bf16(
                        false, af, false, bfr, (short)0, acc[i], false, false);
                }
            }
            #pragma unroll
            for (int i = 0; i < 6; ++i) {
                int c = ((wave + 16 * i) << 4) + arow;
                float bias = b_hh[c];
                #pragma unroll
                for (int v = 0; v < 8; ++v)
                    ghb[(rb + v) * H3_ + c] = acc[i][v] + bias;
            }
        }
        __syncthreads();
        // pass B: gi = x_aug @ w_ih^T + b_ih ; fuse r/z gates
        {
            v8f acc[6] = {};
            const __bf16* Ap = xb + arow * F_ + kh;
            const __bf16* Bbase = WIH + (size_t)arow * F_ + kh;
            #pragma unroll 1
            for (int k0 = 0; k0 < F_; k0 += 32) {
                v16bf af = *(const v16bf*)(Ap + k0);
                #pragma unroll
                for (int i = 0; i < 6; ++i) {
                    int n0 = (wave + 16 * i) << 4;
                    v16bf bfr = *(const v16bf*)(Bbase + (size_t)n0 * F_ + k0);
                    acc[i] = __builtin_amdgcn_wmma_f32_16x16x32_bf16(
                        false, af, false, bfr, (short)0, acc[i], false, false);
                }
            }
            #pragma unroll
            for (int i = 0; i < 6; ++i) {
                int t = wave + 16 * i;           // 0..95
                int c = (t << 4) + arow;         // 0..1535
                float bi = b_ih[c];
                float wlast = w_ih_full[(size_t)c * 129 + 128];  // dt column (k=128)
                #pragma unroll
                for (int v = 0; v < 8; ++v) {
                    int m = rb + v;
                    float g = acc[i][v] + bi + s_dti[m] * wlast;
                    if (t < 32) {                 // r gate
                        float x = g + ghb[m * H3_ + c];
                        rbuf[m * H_ + c] = 1.f / (1.f + expf(-x));
                    } else if (t < 64) {          // z gate
                        float x = g + ghb[m * H3_ + c];
                        zbuf[m * H_ + (c - 512)] = 1.f / (1.f + expf(-x));
                    } else {                      // n pre-activation (gi side)
                        ginb[m * H_ + (c - 1024)] = g;
                    }
                }
            }
        }
        __syncthreads();
        // pass C: n = tanh(gi_n + r*gh_n); h = (1-z)*n + z*h; emit raw output
        for (int e = tid; e < 16 * H_; e += 512) {
            int m = e >> 9, c = e & 511;
            float r = rbuf[e], z = zbuf[e];
            float n = tanhf(ginb[e] + r * ghb[m * H3_ + 1024 + c]);
            float hv = (1.f - z) * n + z * h[e];
            h[e] = hv;
            rbuf[e] = 0.f;              // restore hacc = 0 for next ODE solve
            a_bf[e] = (__bf16)hv;       // keep bf16 A-operand invariant
            out[((size_t)(bt * 16 + m) * S_ + s) * H_ + c] = hv;
        }
        __syncthreads();
    }
}

// ---------------------------------------------------------------------------
// Final LayerNorm over H, in place on d_out: one block per (b,s) row.
// ---------------------------------------------------------------------------
__global__ void layernorm_kernel(float* __restrict__ out,
                                 const float* __restrict__ gamma,
                                 const float* __restrict__ beta) {
    float* p = out + (size_t)blockIdx.x * H_;
    __shared__ float rs[256], rq[256];
    float s0 = 0.f, s1 = 0.f;
    for (int c = threadIdx.x; c < H_; c += 256) {
        float v = p[c]; s0 += v; s1 += v * v;
    }
    rs[threadIdx.x] = s0; rq[threadIdx.x] = s1;
    __syncthreads();
    for (int o = 128; o > 0; o >>= 1) {
        if (threadIdx.x < o) {
            rs[threadIdx.x] += rs[threadIdx.x + o];
            rq[threadIdx.x] += rq[threadIdx.x + o];
        }
        __syncthreads();
    }
    float mean = rs[0] / (float)H_;
    float var  = rq[0] / (float)H_ - mean * mean;
    float inv  = rsqrtf(var + 1e-5f);
    for (int c = threadIdx.x; c < H_; c += 256)
        p[c] = (p[c] - mean) * inv * gamma[c] + beta[c];
}

// ---------------------------------------------------------------------------
extern "C" void kernel_launch(void* const* d_in, const int* in_sizes, int n_in,
                              void* d_out, int out_size, void* d_ws, size_t ws_size,
                              hipStream_t stream) {
    const float* tp     = (const float*)d_in[0];
    const float* values = (const float*)d_in[1];
    const float* w1     = (const float*)d_in[2];
    const float* b1     = (const float*)d_in[3];
    const float* w2     = (const float*)d_in[4];
    const float* b2     = (const float*)d_in[5];
    const float* w_ih   = (const float*)d_in[6];
    const float* w_hh   = (const float*)d_in[7];
    const float* b_ih   = (const float*)d_in[8];
    const float* b_hh   = (const float*)d_in[9];
    const float* gamma  = (const float*)d_in[10];
    const float* beta   = (const float*)d_in[11];
    float* out = (float*)d_out;

    char* ws = (char*)d_ws;
    float*  dtsub = (float*)ws;                              //   512 B
    __bf16* W1T = (__bf16*)(ws + 1024);                      // 1 MiB  [1024][512]
    __bf16* W2T = (__bf16*)(ws + 1024 + 1048576);            // 1 MiB  [512][1024]
    __bf16* WHH = (__bf16*)(ws + 1024 + 2097152);            // 1.5 MiB[1536][512]
    __bf16* WIH = (__bf16*)(ws + 1024 + 2097152 + 1572864);  // 384 KiB[1536][128]

    compute_dt_kernel<<<S_, 256, 0, stream>>>(tp, dtsub);
    // dst[n*K+k] = src[k*sK + n*sN]
    conv_w_kernel<<<1024, 256, 0, stream>>>(w1,   W1T, 512,  1024, 1024, 1);
    conv_w_kernel<<<1024, 256, 0, stream>>>(w2,   W2T, 1024, 512,  512,  1);
    conv_w_kernel<<<1024, 256, 0, stream>>>(w_hh, WHH, 512,  1536, 1,    512);
    conv_w_kernel<<<1024, 256, 0, stream>>>(w_ih, WIH, 128,  1536, 1,    129);

    const int SMEM_BYTES = 278528;  // 272 KB dynamic LDS (<= 320 KB/WGP)
    (void)hipFuncSetAttribute((const void*)ode_rnn_kernel,
                              hipFuncAttributeMaxDynamicSharedMemorySize,
                              SMEM_BYTES);
    ode_rnn_kernel<<<B_ / 16, 512, SMEM_BYTES, stream>>>(
        tp, values, dtsub, W1T, W2T, WHH, WIH, b1, b2, b_ih, b_hh, w_ih, out);

    layernorm_kernel<<<B_ * S_, 256, 0, stream>>>(out, gamma, beta);
    (void)in_sizes; (void)n_in; (void)out_size; (void)ws_size;
}
